// AttentionLayer_4363686772768
// MI455X (gfx1250) — matrix operand order repaired and checked by reading, not compiled, not run on hardware
//
#include <hip/hip_runtime.h>

// ---------------------------------------------------------------------------
// Fused attention (tanh-score additive attention) for MI455X / gfx1250.
//   hidden_states: (N=2048, H=512, S=256) f32 ; W: (H,H) ; bias,c: (H,1)
//   out: (N,H) f32
// One workgroup per batch item, 512 threads (16 wave32).
//  - matmul on matrix cores: v_wmma_f32_16x16x32_f16 (f32 accum)
//  - hidden[n] staged once to LDS (f16, transposed, 260 KB of the 320 KB WGP LDS)
//  - W^T tiles double-buffered via global_load_async_to_lds_b128 (ASYNCcnt)
//  - tanh/softmax epilogue uses v_exp_f32 + v_rcp_f32 (no IEEE divides)
// ---------------------------------------------------------------------------

typedef __attribute__((ext_vector_type(16))) _Float16 v16h;
typedef __attribute__((ext_vector_type(8)))  _Float16 v8h;
typedef __attribute__((ext_vector_type(4)))  _Float16 v4h;
typedef __attribute__((ext_vector_type(8)))  float    v8f;

#define N_BATCH 2048
#define H_DIM   512
#define S_DIM   256
#define AST     520   // padded LDS row stride (halfs): 1040 B/row -> conflict-free halves
#define NT      512   // threads / block = 16 waves

// LDS layout (bytes, dynamic LDS base = 0: no static __shared__ in this kernel)
#define AS_BYTES   (S_DIM * AST * 2)              // 266240: A = hs^T as f16 [s][h]
#define WT0_BYTE   (AS_BYTES)                     // 16 x AST halfs, buffer 0
#define WT_BYTES   (16 * AST * 2)                 // 16640
#define WT1_BYTE   (WT0_BYTE + WT_BYTES)
#define ALN_BYTE   (WT1_BYTE + WT_BYTES)          // 256 f32 alignment/attn
#define LDS_TOTAL  (ALN_BYTE + S_DIM * 4)         // 300544 < 320 KB

__device__ __forceinline__ float tanh_fast(float x) {
  // tanh(x) = 1 - 2/(exp(2x)+1); v_exp_f32 + v_rcp_f32, saturates at +/-inf
  float e = __expf(2.0f * x);
  return 1.0f - 2.0f * __builtin_amdgcn_rcpf(e + 1.0f);
}

// Async DMA: 16 bytes global -> LDS, tracked by ASYNCcnt (CDNA5 path)
__device__ __forceinline__ void async_b128(unsigned lds_byte_off, const void* g) {
  unsigned long long ga = (unsigned long long)(uintptr_t)g;
  asm volatile("global_load_async_to_lds_b128 %0, %1, off"
               :: "v"(lds_byte_off), "v"(ga) : "memory");
}
__device__ __forceinline__ void wait_async0() {
  asm volatile("s_wait_asynccnt 0x0" ::: "memory");
}

// Stage W^T rows j0..j0+15 (f16, contiguous 512 halfs each) into padded LDS tile.
__device__ __forceinline__ void stage_wt_async(const _Float16* __restrict__ Wt,
                                               int j0, unsigned wt_byte_base, int tid) {
  #pragma unroll
  for (int c = 0; c < 2; ++c) {               // 1024 x 16B chunks / 512 threads
    int i    = tid + c * NT;
    int jj   = i >> 6;                        // row in tile
    int part = i & 63;                        // 16B chunk in row
    unsigned lds_off = wt_byte_base + (unsigned)(jj * (AST * 2) + part * 16);
    async_b128(lds_off, (const void*)(Wt + (size_t)j0 * H_DIM + (size_t)i * 8));
  }
}

// One-time prep: W (H,H) f32 row-major [h][j]  ->  Wt f16 [j][h]
__global__ __launch_bounds__(256)
void w_transpose_f16(const float* __restrict__ W, _Float16* __restrict__ Wt) {
  int idx = blockIdx.x * blockDim.x + threadIdx.x;   // over H*H
  int j = idx >> 9;
  int h = idx & (H_DIM - 1);
  Wt[idx] = (_Float16)W[h * H_DIM + j];
}

__global__ __launch_bounds__(NT)
void attn_fused_kernel(const float* __restrict__ hs,     // (N,H,S)
                       const float* __restrict__ bias,   // (H)
                       const float* __restrict__ cvec,   // (H)
                       const _Float16* __restrict__ Wt,  // (H j, H h) f16
                       float* __restrict__ out)          // (N,H)
{
  extern __shared__ char smem_raw[];
  _Float16* As   = (_Float16*)smem_raw;                  // [s][h], padded AST
  _Float16* wt0  = (_Float16*)(smem_raw + WT0_BYTE);
  _Float16* wt1  = (_Float16*)(smem_raw + WT1_BYTE);
  float*    ctx  = (float*)wt0;                          // 4*512 f32 (after matmul)
  float*    aln  = (float*)(smem_raw + ALN_BYTE);        // S_DIM f32

  const int n    = blockIdx.x;
  const int tid  = threadIdx.x;
  const int lane = tid & 31;
  const int wave = tid >> 5;        // 0..15 -> row tile
  const int r    = lane & 15;       // A: row in tile / B: col in tile
  const int half = lane >> 4;       // selects K sub-range per ISA 7.12.2

  const float* hsn = hs + (size_t)n * H_DIM * S_DIM;

  // Kick off async DMA of the first W^T tile before touching anything else.
  stage_wt_async(Wt, 0, WT0_BYTE, tid);

  // ---- Stage hidden[n] (H,S) f32 -> LDS As[s][h] f16 (transpose+convert) ----
  {
    const float4* src = (const float4*)hsn;              // contiguous in s
    for (int i = tid; i < H_DIM * S_DIM / 4; i += NT) {
      int h  = i >> 6;             // S/4 = 64 float4 per h-row
      int s4 = (i & 63) << 2;
      float4 v = src[i];
      _Float16* dst = As + (size_t)s4 * AST + h;
      dst[0]       = (_Float16)v.x;
      dst[AST]     = (_Float16)v.y;
      dst[2 * AST] = (_Float16)v.z;
      dst[3 * AST] = (_Float16)v.w;
    }
  }

  const int s0 = wave * 16;
  float align_acc[8] = {0.f, 0.f, 0.f, 0.f, 0.f, 0.f, 0.f, 0.f};

  // A-frag (16x32 f16): lane half 0 -> K = {0..7,16..23}, half 1 -> {8..15,24..31}
  const _Float16* ap = As + (size_t)(s0 + r) * AST + half * 8;

  // ---- alignment[s] = sum_j tanh( (hs^T @ W)[s,j] + b[j] ) * c[j] ----
  for (int jt = 0; jt < H_DIM / 16; ++jt) {
    wait_async0();        // this wave's DMA into cur buffer landed
    __syncthreads();      // everyone's DMA landed; prev iter's reads complete

    const _Float16* wt = (jt & 1) ? wt1 : wt0;
    if (jt + 1 < H_DIM / 16)  // prefetch next tile into the other buffer
      stage_wt_async(Wt, (jt + 1) * 16, (jt & 1) ? WT0_BYTE : WT1_BYTE, tid);

    const int j0 = jt * 16;
    v8f acc = {};
    // B-frag (32x16 f16): lane half 0 -> K={0..7,16..23}, half 1 -> {8..15,24..31}; N = lane%16
    const _Float16* bp = wt + (size_t)r * AST + half * 8;
    #pragma unroll 4
    for (int kk = 0; kk < H_DIM / 32; ++kk) {
      v8h alo = *(const v8h*)(ap + kk * 32);        // A VGPR0-3
      v8h ahi = *(const v8h*)(ap + kk * 32 + 16);   // A VGPR4-7
      v8h blo = *(const v8h*)(bp + kk * 32);        // B VGPR0-3
      v8h bhi = *(const v8h*)(bp + kk * 32 + 16);   // B VGPR4-7
      v16h a = __builtin_shufflevector(alo, ahi, 0,1,2,3,4,5,6,7,8,9,10,11,12,13,14,15);
      v16h b = __builtin_shufflevector(blo, bhi, 0,1,2,3,4,5,6,7,8,9,10,11,12,13,14,15);
      acc = __builtin_amdgcn_wmma_f32_16x16x32_f16(
          /*neg_a=*/false, a, /*neg_b=*/false, b,
          /*c_mod=*/(short)0, acc, /*reuse_a=*/false, /*reuse_b=*/false);
    }

    // tanh + context-vector weighting on the accumulator fragment
    float bj = bias[j0 + r];   // per-column bias (broadcast over rows)
    float cj = cvec[j0 + r];
    #pragma unroll
    for (int v = 0; v < 8; ++v)
      align_acc[v] += tanh_fast(acc[v] + bj) * cj;
  }

  // ---- reduce over the 16 lanes of each half -> alignment per row ----
  // C layout: VGPR v, lanes 0-15: M=v; lanes 16-31: M=v+8 (N = lane%16)
  #pragma unroll
  for (int v = 0; v < 8; ++v) {
    float x = align_acc[v];
    x += __shfl_xor(x, 1);
    x += __shfl_xor(x, 2);
    x += __shfl_xor(x, 4);
    x += __shfl_xor(x, 8);
    if (r == 0) aln[s0 + half * 8 + v] = x;
  }
  __syncthreads();

  // ---- softmax over S=256 (single wave) ----
  if (tid < 32) {
    float vals[8];
    float m = -INFINITY;
    #pragma unroll
    for (int i = 0; i < 8; ++i) { vals[i] = aln[tid * 8 + i]; m = fmaxf(m, vals[i]); }
    for (int off = 1; off < 32; off <<= 1) m = fmaxf(m, __shfl_xor(m, off));
    float ssum = 0.f;
    #pragma unroll
    for (int i = 0; i < 8; ++i) { vals[i] = __expf(vals[i] - m); ssum += vals[i]; }
    for (int off = 1; off < 32; off <<= 1) ssum += __shfl_xor(ssum, off);
    float inv = __builtin_amdgcn_rcpf(ssum);
    #pragma unroll
    for (int i = 0; i < 8; ++i) aln[tid * 8 + i] = vals[i] * inv;
  }
  __syncthreads();

  // ---- context[h] = sum_s attn[s] * As[s][h]  (reuse staged LDS slab) ----
  {
    int q  = tid >> 7;              // s-quarter 0..3 (64 s each)
    int hb = (tid & 127) << 2;      // 4 consecutive h per thread
    float a0 = 0.f, a1 = 0.f, a2 = 0.f, a3 = 0.f;
    int sBeg = q * 64;
    #pragma unroll 4
    for (int s = sBeg; s < sBeg + 64; ++s) {
      float w = aln[s];
      v4h hv = *(const v4h*)(As + (size_t)s * AST + hb);
      a0 += w * (float)hv[0];
      a1 += w * (float)hv[1];
      a2 += w * (float)hv[2];
      a3 += w * (float)hv[3];
    }
    ctx[q * H_DIM + hb + 0] = a0;
    ctx[q * H_DIM + hb + 1] = a1;
    ctx[q * H_DIM + hb + 2] = a2;
    ctx[q * H_DIM + hb + 3] = a3;
  }
  __syncthreads();
  {
    float rsum = ctx[tid] + ctx[H_DIM + tid] + ctx[2 * H_DIM + tid] + ctx[3 * H_DIM + tid];
    out[(size_t)n * H_DIM + tid] = rsum;
  }
}

extern "C" void kernel_launch(void* const* d_in, const int* in_sizes, int n_in,
                              void* d_out, int out_size, void* d_ws, size_t ws_size,
                              hipStream_t stream) {
  const float* hs   = (const float*)d_in[0];   // (N,H,S)
  const float* W    = (const float*)d_in[1];   // (H,H)
  const float* bias = (const float*)d_in[2];   // (H,1)
  const float* cvec = (const float*)d_in[3];   // (H,1)
  float* out        = (float*)d_out;
  _Float16* Wt      = (_Float16*)d_ws;         // 512 KB: W^T in f16

  // one-time W transpose+convert (L2-resident thereafter)
  w_transpose_f16<<<(H_DIM * H_DIM) / 256, 256, 0, stream>>>(W, Wt);

  attn_fused_kernel<<<N_BATCH, NT, LDS_TOTAL, stream>>>(hs, bias, cvec, Wt, out);
}
